// TopKAttentionMIL_16329465660223
// MI455X (gfx1250) — matrix operand order repaired and checked by reading, not compiled, not run on hardware
//
#include <hip/hip_runtime.h>
#include <math.h>

// ---------------------------------------------------------------------------
// TopK Attention MIL for MI455X (gfx1250, wave32, WMMA)
//
// scores = tanh(x @ Wa1 + ba1) @ Wa2 + ba2      x:[100000,1024] Wa1:[1024,128]
//   -> dominant cost: 410 MB of x @ 23.3 TB/s  (~17.6 us floor)
//   -> 26.2 GFLOP needs ~1.5 PFLOP/s to stay BW-bound => bf16 WMMA, f32 accum.
// Each wave32 owns 32 rows (two 16x16x32 A tiles sharing every B tile load).
// Wa1 is pre-packed once into the ISA B-operand lane layout and stays in L2.
// ---------------------------------------------------------------------------

typedef __attribute__((ext_vector_type(16))) __bf16 v16bf;
typedef __attribute__((ext_vector_type(8)))  float  v8f;
typedef __attribute__((ext_vector_type(4)))  float  f32x4;   // clang vector: OK for nontemporal builtins

#define N_ROWS   100000
#define DIM      1024
#define ADIM     128
#define HDIM     256
#define KTOP     16
#define TILES32  (N_ROWS / 32)            // 3125 row tiles of 32
#define WAVES_PB 8                        // 256 threads = 8 wave32
#define NBLK1    128                      // stage-1 top-k blocks
#define SEG      ((N_ROWS + NBLK1 - 1) / NBLK1)   // 782 elems per block
#define NCAND    (NBLK1 * KTOP)           // 2048 candidates

// output layout (floats): [logit(1) | slide_embedding(1024) | full_weights(100000) | top_idx(16)]
#define OUT_SE   1
#define OUT_W    (1 + DIM)
#define OUT_IDX  (1 + DIM + N_ROWS)

// workspace layout (bytes)
#define WS_PACKED_OFF 0                          // 131072 bf16 (as u16) = 262144 B
#define WS_SCORES_OFF 262144                     // 100000 f32 = 400000 B
#define WS_CVAL_OFF   (262144 + 400000)          // 2048 f32
#define WS_CIDX_OFF   (WS_CVAL_OFF + NCAND * 4)  // 2048 i32

// Native bf16 conversion: lets clang emit the packed hardware converter
// (v_cvt_pk_bf16_f32-class) instead of scalar bfe/add3 rounding chains.
__device__ __forceinline__ __bf16 cvt_bf16(float f) {
  return static_cast<__bf16>(f);
}
__device__ __forceinline__ unsigned short f32_to_bf16_bits(float f) {
  return __builtin_bit_cast(unsigned short, static_cast<__bf16>(f));
}

// ---------------------------------------------------------------------------
// Kernel 1: pack Wa1 [1024x128] f32 into bf16 WMMA B-operand tiles.
// Tile t = kc*8 + nc covers K rows kc*32..kc*32+31, cols nc*16..nc*16+15.
// Element index within tile = lane*16 + j, where lane%16 = column,
// lanes 0-15 hold K 0..7,16..23, lanes 16-31 hold K 8..15,24..31 (ISA B layout).
// ---------------------------------------------------------------------------
__global__ void __launch_bounds__(256)
pack_wa1_kernel(const float* __restrict__ Wa1, unsigned short* __restrict__ pk) {
  int idx = blockIdx.x * 256 + threadIdx.x;      // 0 .. 131071
  int tile   = idx >> 9;                         // /512
  int within = idx & 511;
  int lane = within >> 4;
  int j    = within & 15;
  int kc = tile >> 3;
  int nc = tile & 7;
  int n  = lane & 15;
  int k  = kc * 32 + ((lane >> 4) << 3) + j + ((j >> 3) << 3);  // j<8: 8h+j, j>=8: 16+8h+(j-8)
  pk[idx] = f32_to_bf16_bits(Wa1[k * ADIM + nc * 16 + n]);
}

// ---------------------------------------------------------------------------
// Kernel 2: attention scores. One wave32 per 32-row tile (two A tiles).
// Every packed-B tile load feeds two v_wmma_f32_16x16x32_bf16 ops.
// x is streamed with non-temporal loads so Wa1 stays L2-resident.
// ---------------------------------------------------------------------------
__global__ void __launch_bounds__(256)
scores_kernel(const float* __restrict__ x, const unsigned short* __restrict__ pk,
              const float* __restrict__ ba1, const float* __restrict__ Wa2,
              const float* __restrict__ ba2, float* __restrict__ scores) {
  const int tile = blockIdx.x * WAVES_PB + (threadIdx.x >> 5);
  if (tile >= TILES32) return;                   // wave-uniform: EXEC stays all-1s
  const int lane  = threadIdx.x & 31;
  const int col   = lane & 15;
  const int base  = tile * 32;
  const int khalf = (lane >> 4) << 3;            // 0 or 8 (A-layout half-wave split)
  const float* xrow0 = x + (size_t)(base + col) * DIM;
  const float* xrow1 = x + (size_t)(base + 16 + col) * DIM;

  float b1[8], w2[8];
#pragma unroll
  for (int nc = 0; nc < 8; ++nc) {
    b1[nc] = ba1[nc * 16 + col];
    w2[nc] = Wa2[nc * 16 + col];
  }

  v8f acc0[8] = {};
  v8f acc1[8] = {};

  for (int kc = 0; kc < 32; ++kc) {
    const int k0 = kc * 32 + khalf;
    f32x4 r0c0 = __builtin_nontemporal_load((const f32x4*)(xrow0 + k0));
    f32x4 r0c1 = __builtin_nontemporal_load((const f32x4*)(xrow0 + k0 + 4));
    f32x4 r0c2 = __builtin_nontemporal_load((const f32x4*)(xrow0 + k0 + 16));
    f32x4 r0c3 = __builtin_nontemporal_load((const f32x4*)(xrow0 + k0 + 20));
    f32x4 r1c0 = __builtin_nontemporal_load((const f32x4*)(xrow1 + k0));
    f32x4 r1c1 = __builtin_nontemporal_load((const f32x4*)(xrow1 + k0 + 4));
    f32x4 r1c2 = __builtin_nontemporal_load((const f32x4*)(xrow1 + k0 + 16));
    f32x4 r1c3 = __builtin_nontemporal_load((const f32x4*)(xrow1 + k0 + 20));
    __builtin_prefetch(xrow0 + k0 + 32, 0, 0);   // global_prefetch next K chunk
    __builtin_prefetch(xrow1 + k0 + 32, 0, 0);

    v16bf a0, a1;
    a0[0]  = cvt_bf16(r0c0.x); a0[1]  = cvt_bf16(r0c0.y);
    a0[2]  = cvt_bf16(r0c0.z); a0[3]  = cvt_bf16(r0c0.w);
    a0[4]  = cvt_bf16(r0c1.x); a0[5]  = cvt_bf16(r0c1.y);
    a0[6]  = cvt_bf16(r0c1.z); a0[7]  = cvt_bf16(r0c1.w);
    a0[8]  = cvt_bf16(r0c2.x); a0[9]  = cvt_bf16(r0c2.y);
    a0[10] = cvt_bf16(r0c2.z); a0[11] = cvt_bf16(r0c2.w);
    a0[12] = cvt_bf16(r0c3.x); a0[13] = cvt_bf16(r0c3.y);
    a0[14] = cvt_bf16(r0c3.z); a0[15] = cvt_bf16(r0c3.w);
    a1[0]  = cvt_bf16(r1c0.x); a1[1]  = cvt_bf16(r1c0.y);
    a1[2]  = cvt_bf16(r1c0.z); a1[3]  = cvt_bf16(r1c0.w);
    a1[4]  = cvt_bf16(r1c1.x); a1[5]  = cvt_bf16(r1c1.y);
    a1[6]  = cvt_bf16(r1c1.z); a1[7]  = cvt_bf16(r1c1.w);
    a1[8]  = cvt_bf16(r1c2.x); a1[9]  = cvt_bf16(r1c2.y);
    a1[10] = cvt_bf16(r1c2.z); a1[11] = cvt_bf16(r1c2.w);
    a1[12] = cvt_bf16(r1c3.x); a1[13] = cvt_bf16(r1c3.y);
    a1[14] = cvt_bf16(r1c3.z); a1[15] = cvt_bf16(r1c3.w);

    const unsigned short* pt = pk + (size_t)kc * (8 * 512) + (size_t)lane * 16;
#pragma unroll
    for (int nc = 0; nc < 8; ++nc) {
      v16bf b = *(const v16bf*)(pt + nc * 512);  // 32B/lane, contiguous tile
      acc0[nc] = __builtin_amdgcn_wmma_f32_16x16x32_bf16(
          false, a0, false, b, (short)0, acc0[nc], false, false);
      acc1[nc] = __builtin_amdgcn_wmma_f32_16x16x32_bf16(
          false, a1, false, b, (short)0, acc1[nc], false, false);
    }
  }

  const float bias2 = ba2[0];
#pragma unroll
  for (int r = 0; r < 8; ++r) {
    float s0 = 0.0f, s1 = 0.0f;
#pragma unroll
    for (int nc = 0; nc < 8; ++nc) {
      s0 += tanhf(acc0[nc][r] + b1[nc]) * w2[nc];
      s1 += tanhf(acc1[nc][r] + b1[nc]) * w2[nc];
    }
    // reduce across the 16 lanes of each half-wave (masks < 16 keep halves apart)
    s0 += __shfl_xor(s0, 1); s1 += __shfl_xor(s1, 1);
    s0 += __shfl_xor(s0, 2); s1 += __shfl_xor(s1, 2);
    s0 += __shfl_xor(s0, 4); s1 += __shfl_xor(s1, 4);
    s0 += __shfl_xor(s0, 8); s1 += __shfl_xor(s1, 8);
    if (lane == 0) {
      scores[base + r]          = s0 + bias2;    // tile-A rows 0..7
      scores[base + 16 + r]     = s1 + bias2;    // tile-B rows 0..7
    }
    if (lane == 16) {
      scores[base + 8 + r]      = s0 + bias2;    // tile-A rows 8..15
      scores[base + 24 + r]     = s1 + bias2;    // tile-B rows 8..15
    }
  }
}

// ---------------------------------------------------------------------------
// Kernel 3: per-block top-16 candidates (iterative argmax over a segment).
// ---------------------------------------------------------------------------
__global__ void __launch_bounds__(256)
topk_stage1_kernel(const float* __restrict__ scores,
                   float* __restrict__ cval, int* __restrict__ cidx) {
  __shared__ float sv[256];
  __shared__ int   si[256];
  const int tid  = threadIdx.x;
  const int base = blockIdx.x * SEG;
  const float NEG_INF = -__builtin_inff();

  float v[4];
  int   gi[4];
#pragma unroll
  for (int s = 0; s < 4; ++s) {
    int loc = s * 256 + tid;
    int g   = base + loc;
    bool ok = (loc < SEG) && (g < N_ROWS);
    v[s]  = ok ? scores[g] : NEG_INF;
    gi[s] = ok ? g : -1;
  }

  for (int r = 0; r < KTOP; ++r) {
    float bv = v[0]; int bi = gi[0];
#pragma unroll
    for (int s = 1; s < 4; ++s)
      if (v[s] > bv) { bv = v[s]; bi = gi[s]; }
    sv[tid] = bv; si[tid] = bi;
    __syncthreads();
    for (int st = 128; st > 0; st >>= 1) {
      if (tid < st && sv[tid + st] > sv[tid]) { sv[tid] = sv[tid + st]; si[tid] = si[tid + st]; }
      __syncthreads();
    }
    int   wi = si[0];
    float wv = sv[0];
    if (tid == 0) {
      cval[blockIdx.x * KTOP + r] = wv;
      cidx[blockIdx.x * KTOP + r] = wi;
    }
#pragma unroll
    for (int s = 0; s < 4; ++s)
      if (gi[s] == wi) v[s] = NEG_INF;
    __syncthreads();
  }
}

// ---------------------------------------------------------------------------
// Kernel 4: zero the full_weights output region.
// ---------------------------------------------------------------------------
__global__ void __launch_bounds__(256)
zero_weights_kernel(float* __restrict__ out) {
  int i = blockIdx.x * 256 + threadIdx.x;
  if (i < N_ROWS) out[OUT_W + i] = 0.0f;
}

// ---------------------------------------------------------------------------
// Kernel 5: merge candidates -> global top-16, softmax, weighted gather,
// classifier MLP, scatter, all output writes. Single block of 256.
// ---------------------------------------------------------------------------
__global__ void __launch_bounds__(256)
finalize_kernel(const float* __restrict__ x,
                const float* __restrict__ Wc1, const float* __restrict__ bc1,
                const float* __restrict__ Wc2, const float* __restrict__ bc2,
                const float* __restrict__ cval, const int* __restrict__ cidx,
                float* __restrict__ out) {
  __shared__ float cv[NCAND];
  __shared__ int   ci[NCAND];
  __shared__ float rv[256];
  __shared__ int   rp[256];
  __shared__ float topv[KTOP];
  __shared__ int   topi[KTOP];
  __shared__ float attnw[KTOP];
  __shared__ float se[DIM];
  const int tid = threadIdx.x;
  const float NEG_INF = -__builtin_inff();

#pragma unroll
  for (int s = 0; s < NCAND / 256; ++s) {
    cv[s * 256 + tid] = cval[s * 256 + tid];
    ci[s * 256 + tid] = cidx[s * 256 + tid];
  }
  __syncthreads();

  for (int r = 0; r < KTOP; ++r) {
    float bv = NEG_INF; int bp = 0;
#pragma unroll
    for (int s = 0; s < NCAND / 256; ++s) {
      int p = s * 256 + tid;
      if (cv[p] > bv) { bv = cv[p]; bp = p; }
    }
    rv[tid] = bv; rp[tid] = bp;
    __syncthreads();
    for (int st = 128; st > 0; st >>= 1) {
      if (tid < st && rv[tid + st] > rv[tid]) { rv[tid] = rv[tid + st]; rp[tid] = rp[tid + st]; }
      __syncthreads();
    }
    if (tid == 0) {
      int p   = rp[0];
      topv[r] = rv[0];
      topi[r] = ci[p];
      cv[p]   = NEG_INF;
    }
    __syncthreads();
  }

  // softmax over the 16 kept scores
  if (tid == 0) {
    float mx = topv[0];
    for (int k = 1; k < KTOP; ++k) mx = fmaxf(mx, topv[k]);
    float sum = 0.0f;
    for (int k = 0; k < KTOP; ++k) { float e = expf(topv[k] - mx); attnw[k] = e; sum += e; }
    float inv = 1.0f / sum;
    for (int k = 0; k < KTOP; ++k) attnw[k] *= inv;
  }
  __syncthreads();

  // scatter attention weights + emit indices
  if (tid < KTOP) {
    out[OUT_W + topi[tid]]  = attnw[tid];
    out[OUT_IDX + tid]      = (float)topi[tid];
  }

  // slide embedding = attn_w @ x[top_idx]
  for (int d = tid; d < DIM; d += 256) {
    float s = 0.0f;
#pragma unroll
    for (int k = 0; k < KTOP; ++k)
      s += attnw[k] * x[(size_t)topi[k] * DIM + d];
    se[d] = s;
    out[OUT_SE + d] = s;
  }
  __syncthreads();

  // classifier: h = relu(se @ Wc1 + bc1); logit = h @ Wc2 + bc2
  float hj = bc1[tid];
  for (int d = 0; d < DIM; ++d)
    hj += se[d] * Wc1[d * HDIM + tid];            // coalesced across tid
  hj = fmaxf(hj, 0.0f);
  rv[tid] = hj * Wc2[tid];
  __syncthreads();
  for (int st = 128; st > 0; st >>= 1) {
    if (tid < st) rv[tid] += rv[tid + st];
    __syncthreads();
  }
  if (tid == 0) out[0] = rv[0] + bc2[0];
}

// ---------------------------------------------------------------------------
extern "C" void kernel_launch(void* const* d_in, const int* in_sizes, int n_in,
                              void* d_out, int out_size, void* d_ws, size_t ws_size,
                              hipStream_t stream) {
  (void)in_sizes; (void)n_in; (void)out_size; (void)ws_size;
  const float* x   = (const float*)d_in[0];
  const float* Wa1 = (const float*)d_in[1];
  const float* ba1 = (const float*)d_in[2];
  const float* Wa2 = (const float*)d_in[3];
  const float* ba2 = (const float*)d_in[4];
  const float* Wc1 = (const float*)d_in[5];
  const float* bc1 = (const float*)d_in[6];
  const float* Wc2 = (const float*)d_in[7];
  const float* bc2 = (const float*)d_in[8];
  float* out = (float*)d_out;

  char* ws = (char*)d_ws;
  unsigned short* pk = (unsigned short*)(ws + WS_PACKED_OFF);
  float* scores = (float*)(ws + WS_SCORES_OFF);
  float* cval   = (float*)(ws + WS_CVAL_OFF);
  int*   cidx   = (int*)(ws + WS_CIDX_OFF);

  pack_wa1_kernel<<<(DIM * ADIM) / 256, 256, 0, stream>>>(Wa1, pk);
  scores_kernel<<<(TILES32 + WAVES_PB - 1) / WAVES_PB, 256, 0, stream>>>(
      x, pk, ba1, Wa2, ba2, scores);
  topk_stage1_kernel<<<NBLK1, 256, 0, stream>>>(scores, cval, cidx);
  zero_weights_kernel<<<(N_ROWS + 255) / 256, 256, 0, stream>>>(out);
  finalize_kernel<<<1, 256, 0, stream>>>(x, Wc1, bc1, Wc2, bc2, cval, cidx, out);
}
